// DeforAttFusion_74904229642682
// MI455X (gfx1250) — compile-verified, matrix operand order checked by reading, not run on hardware
//
#include <hip/hip_runtime.h>
#include <hip/hip_bf16.h>

typedef __attribute__((ext_vector_type(16))) _Float16 v16h;
typedef __attribute__((ext_vector_type(8)))  float    v8f;

#define C_DIM 256
#define H_DIM 100
#define W_DIM 252
#define HW_DIM (H_DIM * W_DIM)          // 25200
#define N_LVL 2                          // MAX_LEVELS == record_len values
#define N_PTS 4
#define M_TILES (HW_DIM / 16)            // 1575, exact
#define KB_CNT (C_DIM / 32)              // 8 K-steps of 32
#define WARPED_ELEMS ((size_t)2 * N_LVL * C_DIM * HW_DIM)   // b * l * C * HW
#define LVL_STRIDE (C_DIM * HW_DIM)      // plane distance level0 -> level1 (same c)

// ---------------------------------------------------------------------------
// Stage 1: affine warp (align_corners=True, zero padding) into workspace,
// channel-first layout warped[(bt*2+l)*C + c][hw]. Thread = output pixel,
// loop over channels: reads along W are coalesced-ish, writes coalesced.
// ---------------------------------------------------------------------------
__global__ void warp_affine_kernel(const float* __restrict__ x,
                                   const int*   __restrict__ rl,
                                   const float* __restrict__ T,
                                   float* __restrict__ warped) {
    int p  = blockIdx.x * blockDim.x + threadIdx.x;
    int l  = blockIdx.y;
    int bt = blockIdx.z;
    if (p >= HW_DIM) return;
    int n = rl[bt];
    if (l >= n) return;
    int start = 0;
    for (int i = 0; i < bt; ++i) start += rl[i];

    const float* img = x + (size_t)(start + l) * C_DIM * HW_DIM;
    // pairwise_t_matrix[bt][0][l][i][j], shape (B,2,2,2,3)
    const float* M = T + (((size_t)bt * 2 + 0) * 2 + l) * 6;

    int py = p / W_DIM, px = p % W_DIM;
    float gx = -1.f + 2.f * (float)px / (float)(W_DIM - 1);
    float gy = -1.f + 2.f * (float)py / (float)(H_DIM - 1);
    float sx = M[0] * gx + M[1] * gy + M[2];
    float sy = M[3] * gx + M[4] * gy + M[5];
    // align_corners = True
    float ix = (sx + 1.f) * 0.5f * (float)(W_DIM - 1);
    float iy = (sy + 1.f) * 0.5f * (float)(H_DIM - 1);
    float x0f = floorf(ix), y0f = floorf(iy);
    int   x0  = (int)x0f,   y0  = (int)y0f;
    float wx1 = ix - x0f,   wy1 = iy - y0f;
    float wx0 = 1.f - wx1,  wy0 = 1.f - wy1;

    int  x1 = x0 + 1, y1 = y0 + 1;
    bool vx0 = (x0 >= 0) & (x0 < W_DIM), vx1 = (x1 >= 0) & (x1 < W_DIM);
    bool vy0 = (y0 >= 0) & (y0 < H_DIM), vy1 = (y1 >= 0) & (y1 < H_DIM);
    int cx0 = min(max(x0, 0), W_DIM - 1), cx1 = min(max(x1, 0), W_DIM - 1);
    int cy0 = min(max(y0, 0), H_DIM - 1), cy1 = min(max(y1, 0), H_DIM - 1);
    float w00 = wx0 * wy0 * (float)(vx0 & vy0);
    float w10 = wx1 * wy0 * (float)(vx1 & vy0);
    float w01 = wx0 * wy1 * (float)(vx0 & vy1);
    float w11 = wx1 * wy1 * (float)(vx1 & vy1);
    int o00 = cy0 * W_DIM + cx0, o10 = cy0 * W_DIM + cx1;
    int o01 = cy1 * W_DIM + cx0, o11 = cy1 * W_DIM + cx1;

    float* dst = warped + (size_t)(bt * N_LVL + l) * C_DIM * HW_DIM + p;
#pragma unroll 4
    for (int c = 0; c < C_DIM; ++c) {
        const float* pl = img + (size_t)c * HW_DIM;
        float v = w00 * pl[o00] + w10 * pl[o10] + w01 * pl[o01] + w11 * pl[o11];
        dst[(size_t)c * HW_DIM] = v;
    }
}

// ---------------------------------------------------------------------------
// Stage 2: WMMA projection GEMM.
// q[hw][c] = warped[bt][0][c][hw];  out[hw][0:16] = q @ w_off + b_off,
//                                   out[hw][16:24] = q @ w_attn + b_attn.
// One wave per 16-row M-tile, 2 N-tiles. B is staged once per block in LDS
// in WMMA fragment layout (f16), branch-free; K-loop fully unrolled.
// ---------------------------------------------------------------------------
__global__ void proj_wmma_kernel(const float* __restrict__ warped,
                                 const float* __restrict__ w_off,
                                 const float* __restrict__ b_off,
                                 const float* __restrict__ w_attn,
                                 const float* __restrict__ b_attn,
                                 float* __restrict__ offattn) {
    // [tile][kb][lane] fragments, 2*8*32 * 32B = 16 KB LDS
    __shared__ v16h lbs[2][KB_CNT][32];

    int tid  = threadIdx.x;
    int wave = blockIdx.x * (blockDim.x >> 5) + (tid >> 5);
    int lane = tid & 31;
    int bt   = blockIdx.y;

    // Cooperative branch-free fill of B fragments (f32 -> f16 convert once).
    {
        _Float16* lp = (_Float16*)lbs;
        for (int idx = tid; idx < 2 * KB_CNT * 32 * 16; idx += blockDim.x) {
            int i  = idx & 15;
            int ln = (idx >> 4) & 31;
            int kb = (idx >> 9) & 7;
            int t  = idx >> 12;
            int mm = ln & 15;
            int kk = kb * 32 + ((ln >> 4) ? 8 : 0) + ((i < 8) ? i : i + 8);
            // t==0: w_off col mm ; t==1: w_attn col mm (mm<8) else 0
            float v0 = w_off[kk * 16 + mm];
            float v1 = w_attn[kk * 8 + (mm & 7)] * (float)(mm < 8);
            lp[idx] = (_Float16)((t == 0) ? v0 : v1);
        }
    }
    __syncthreads();

    if (wave >= M_TILES) return;           // uniform per wave, after the barrier
    int row0  = wave * 16;
    int m     = lane & 15;                 // A: row; C/D: column
    int kbase = (lane >> 4) ? 8 : 0;       // 16-bit operand K-split across lane halves

    const float* q = warped + (size_t)(bt * N_LVL + 0) * C_DIM * HW_DIM + row0 + m;

    v8f acc0 = {}, acc1 = {};
#pragma unroll
    for (int kb = 0; kb < KB_CNT; ++kb) {
        v16h af;
#pragma unroll
        for (int i = 0; i < 16; ++i) {
            // 16-bit A 16x32 layout: element i -> K = (i<8 ? i : i+8) + kbase
            int kk = kb * 32 + kbase + ((i < 8) ? i : i + 8);
            af[i] = (_Float16)q[(size_t)kk * HW_DIM];
        }
        v16h bf0 = lbs[0][kb][lane];
        v16h bf1 = lbs[1][kb][lane];
        acc0 = __builtin_amdgcn_wmma_f32_16x16x32_f16(false, af, false, bf0,
                                                      (short)0, acc0, false, false);
        acc1 = __builtin_amdgcn_wmma_f32_16x16x32_f16(false, af, false, bf1,
                                                      (short)0, acc1, false, false);
    }

    // C/D layout: VGPR r -> M = r (lanes 0-15) / r+8 (lanes 16-31), N = lane&15
    int mofs = (lane >> 4) ? 8 : 0;
#pragma unroll
    for (int r = 0; r < 8; ++r) {
        int hw = row0 + r + mofs;
        float* dst = offattn + ((size_t)bt * HW_DIM + hw) * 24;
        dst[m] = acc0[r] + b_off[m];
        if (m < 8) dst[16 + m] = acc1[r] + b_attn[m];
    }
}

// ---------------------------------------------------------------------------
// Stage 3: per-query softmax over 8 attention logits + 8-point deformable
// bilinear sampling (align_corners=False, zero padding), weighted channel sum.
// Level displacement folded into integer corner offsets -> single base pointer
// in the channel loop. warped stays L2-resident.
// ---------------------------------------------------------------------------
__global__ void defatt_sample_kernel(const float* __restrict__ warped,
                                     const float* __restrict__ offattn,
                                     float* __restrict__ out) {
    int qi = blockIdx.x * blockDim.x + threadIdx.x;
    int bt = blockIdx.y;
    if (qi >= HW_DIM) return;

    const float* oa = offattn + ((size_t)bt * HW_DIM + qi) * 24;

    // softmax over the 8 logits (n * NUM_POINTS = 8)
    float lg[8], mx = -1e30f;
#pragma unroll
    for (int i = 0; i < 8; ++i) { lg[i] = oa[16 + i]; mx = fmaxf(mx, lg[i]); }
    float se = 0.f;
#pragma unroll
    for (int i = 0; i < 8; ++i) { lg[i] = __expf(lg[i] - mx); se += lg[i]; }
    float inv = 1.f / se;

    int py = qi / W_DIM, px = qi % W_DIM;
    float refx = ((float)px + 0.5f) / (float)W_DIM;
    float refy = ((float)py + 0.5f) / (float)H_DIM;

    int   so00[8], so10[8], so01[8], so11[8];
    float sw00[8], sw10[8], sw01[8], sw11[8];
#pragma unroll
    for (int s = 0; s < 8; ++s) {
        float offx = oa[s * 2 + 0], offy = oa[s * 2 + 1];
        float locx = refx + offx / (float)W_DIM;
        float locy = refy + offy / (float)H_DIM;
        float gx = 2.f * locx - 1.f, gy = 2.f * locy - 1.f;
        // align_corners = False
        float ix = ((gx + 1.f) * (float)W_DIM - 1.f) * 0.5f;
        float iy = ((gy + 1.f) * (float)H_DIM - 1.f) * 0.5f;
        float x0f = floorf(ix), y0f = floorf(iy);
        int   x0  = (int)x0f,   y0  = (int)y0f;
        float wx1 = ix - x0f,   wy1 = iy - y0f;
        float wx0 = 1.f - wx1,  wy0 = 1.f - wy1;
        int  x1 = x0 + 1, y1 = y0 + 1;
        bool vx0 = (x0 >= 0) & (x0 < W_DIM), vx1 = (x1 >= 0) & (x1 < W_DIM);
        bool vy0 = (y0 >= 0) & (y0 < H_DIM), vy1 = (y1 >= 0) & (y1 < H_DIM);
        int cx0 = min(max(x0, 0), W_DIM - 1), cx1 = min(max(x1, 0), W_DIM - 1);
        int cy0 = min(max(y0, 0), H_DIM - 1), cy1 = min(max(y1, 0), H_DIM - 1);
        float aw = lg[s] * inv;
        sw00[s] = aw * wx0 * wy0 * (float)(vx0 & vy0);
        sw10[s] = aw * wx1 * wy0 * (float)(vx1 & vy0);
        sw01[s] = aw * wx0 * wy1 * (float)(vx0 & vy1);
        sw11[s] = aw * wx1 * wy1 * (float)(vx1 & vy1);
        int lshift = (s < 4) ? 0 : LVL_STRIDE;   // fold level-1 plane displacement
        so00[s] = cy0 * W_DIM + cx0 + lshift;  so10[s] = cy0 * W_DIM + cx1 + lshift;
        so01[s] = cy1 * W_DIM + cx0 + lshift;  so11[s] = cy1 * W_DIM + cx1 + lshift;
    }

    const float* wp0 = warped + (size_t)(bt * N_LVL + 0) * C_DIM * HW_DIM;
    float* dst = out + (size_t)bt * C_DIM * HW_DIM + qi;
#pragma unroll 2
    for (int c = 0; c < C_DIM; ++c) {
        const float* p0 = wp0 + (size_t)c * HW_DIM;
        float acc = 0.f;
#pragma unroll
        for (int s = 0; s < 8; ++s) {
            acc += sw00[s] * p0[so00[s]];
            acc += sw10[s] * p0[so10[s]];
            acc += sw01[s] * p0[so01[s]];
            acc += sw11[s] * p0[so11[s]];
        }
        dst[(size_t)c * HW_DIM] = acc;
    }
}

// ---------------------------------------------------------------------------
extern "C" void kernel_launch(void* const* d_in, const int* in_sizes, int n_in,
                              void* d_out, int out_size, void* d_ws, size_t ws_size,
                              hipStream_t stream) {
    const float* x      = (const float*)d_in[0];
    const int*   rl     = (const int*)  d_in[1];
    const float* T      = (const float*)d_in[2];
    const float* w_off  = (const float*)d_in[3];
    const float* b_off  = (const float*)d_in[4];
    const float* w_attn = (const float*)d_in[5];
    const float* b_attn = (const float*)d_in[6];
    float* out = (float*)d_out;

    float* warped  = (float*)d_ws;                 // 2*2*256*25200 floats (~103 MB)
    float* offattn = warped + WARPED_ELEMS;        // 2*25200*24 floats

    dim3 g1((HW_DIM + 255) / 256, N_LVL, 2);
    warp_affine_kernel<<<g1, 256, 0, stream>>>(x, rl, T, warped);

    dim3 g2((M_TILES + 7) / 8, 2);                 // 8 waves (tiles) per block
    proj_wmma_kernel<<<g2, 256, 0, stream>>>(warped, w_off, b_off, w_attn, b_attn, offattn);

    dim3 g3((HW_DIM + 255) / 256, 2);
    defatt_sample_kernel<<<g3, 256, 0, stream>>>(warped, offattn, out);
}
